// BevDeformableTemporalAttention_82884278879186
// MI455X (gfx1250) — matrix operand order repaired and checked by reading, not compiled, not run on hardware
//
#include <hip/hip_runtime.h>
#include <hip/hip_bf16.h>

// ---------------------------------------------------------------------------
// BEV deformable temporal attention for gfx1250 (MI455X), wave32 + WMMA f16.
// Round 3: 16x64 tiles/wave (no spills), ping-pong double-buffered k-loops.
// ---------------------------------------------------------------------------

#define NQ   40000
#define EDIM 256
#define NH   8
#define DDIM 32
#define BEVH 200
#define BEVW 200
#define CCONV 192        // 128 offset channels + 64 attn channels
#define KCONV 4608       // 9 taps * 512 in-channels

typedef _Float16 v8h  __attribute__((ext_vector_type(8)));
typedef _Float16 v16h __attribute__((ext_vector_type(16)));
typedef float    v8f  __attribute__((ext_vector_type(8)));

// Load 16 f16 for a WMMA fragment as two contiguous 8-f16 chunks.
__device__ __forceinline__ v16h load_frag16(const _Float16* rowp, int off0, int off1) {
  v8h lo = *(const v8h*)(rowp + off0);
  v8h hi = *(const v8h*)(rowp + off1);
  return __builtin_shufflevector(lo, hi, 0,1,2,3,4,5,6,7,8,9,10,11,12,13,14,15);
}

// One pipeline stage: A fragment + 4 B fragments (16x64 tile).
struct Frags {
  v16h a;
  v16h b[4];
};

// ---------------------------------------------------------------------------
// Prep kernels
// ---------------------------------------------------------------------------

// Build f16 GEMM-A for value projection (val = [pre; query], 80000 x 256)
// and f16 conv input q (40000 x 512 = [pre || query+query_pos]).
__global__ void prep_data_kernel(const float* __restrict__ query,
                                 const float* __restrict__ query_pos,
                                 const float* __restrict__ pre_bev,
                                 const unsigned char* __restrict__ sos,
                                 _Float16* __restrict__ val_h,
                                 _Float16* __restrict__ q_h) {
  int i = blockIdx.x * blockDim.x + threadIdx.x;
  if (i >= NQ * EDIM) return;
  int pos = i >> 8;
  int e   = i & 255;
  float qv  = query[i];
  float pre = (sos[0] != 0) ? qv : pre_bev[i];
  val_h[i]             = (_Float16)pre;
  val_h[NQ * EDIM + i] = (_Float16)qv;
  q_h[(size_t)pos * 512 + e]       = (_Float16)pre;
  q_h[(size_t)pos * 512 + 256 + e] = (_Float16)(qv + query_pos[i]);
}

// Wv / Wo -> f16 row-major ([n][k], which is the WMMA B per-lane layout).
__global__ void prep_gemm_w_kernel(const float* __restrict__ Wv,
                                   const float* __restrict__ Wo,
                                   _Float16* __restrict__ Wvh,
                                   _Float16* __restrict__ Woh) {
  int i = blockIdx.x * blockDim.x + threadIdx.x;
  if (i >= EDIM * EDIM) return;
  Wvh[i] = (_Float16)Wv[i];
  Woh[i] = (_Float16)Wo[i];
}

// Conv weights (off_w 128x512x3x3, attn_w 64x512x3x3) -> f16 [oc][tap][ic].
__global__ void prep_conv_w_kernel(const float* __restrict__ off_w,
                                   const float* __restrict__ attn_w,
                                   _Float16* __restrict__ Wc) {
  int i = blockIdx.x * blockDim.x + threadIdx.x;
  if (i >= CCONV * KCONV) return;
  int oc  = i / KCONV;
  int r   = i % KCONV;
  int tap = r / 512;
  int ic  = r % 512;
  float v = (oc < 128) ? off_w [((size_t)(oc       * 512 + ic)) * 9 + tap]
                       : attn_w[((size_t)((oc-128) * 512 + ic)) * 9 + tap];
  Wc[i] = (_Float16)v;
}

// ---------------------------------------------------------------------------
// GEMM: C[M,N] = A[M,K](f16) @ B[n][k](f16)^T + bias (+ residual).
// One wave computes a 16(M) x 64(N) tile; k-loop ping-pong pipelined.
// K must be a multiple of 64 (K = 256 here -> 8 k-steps of 32).
// ---------------------------------------------------------------------------
__global__ __launch_bounds__(128, 1)
void gemm_kernel(const _Float16* __restrict__ A,
                 const _Float16* __restrict__ B,
                 const float* __restrict__ bias,
                 const float* __restrict__ residual,
                 float* __restrict__ C,
                 int M, int K, int N) {
  int wave = blockIdx.x * (blockDim.x >> 5) + (threadIdx.x >> 5);
  int lane = threadIdx.x & 31;
  int ngroups = N >> 6;
  int mt = wave / ngroups, ng = wave % ngroups;
  int m0 = mt * 16;
  if (m0 >= M) return;
  int n0 = ng * 64;
  int lm    = lane & 15;
  int aoff  = (lane >> 4) << 4;   // 0 or 16
  int boffb = (lane >> 4) << 4;   // 0 or 16

  const _Float16* ap  = A + (size_t)(m0 + lm) * K + aoff;
  const _Float16* bp0 = B + (size_t)(n0 + lm) * K + boffb;

  // Warm the weight rows (global_prefetch_b8).
#pragma unroll
  for (int j = 0; j < 4; ++j)
    __builtin_prefetch(bp0 + (size_t)j * 16 * K, 0, 0);

  auto load_stage = [&](Frags& f, int k) {
    f.a = load_frag16(ap, k, k + 8);
#pragma unroll
    for (int j = 0; j < 4; ++j)
      f.b[j] = load_frag16(bp0 + (size_t)j * 16 * K, k, k + 8);
  };

  v8f acc[4] = {};
  Frags f0, f1;
  load_stage(f0, 0);

  for (int k0 = 0; k0 < K; k0 += 64) {
    load_stage(f1, k0 + 32);            // always in range (K % 64 == 0)
#pragma unroll
    for (int j = 0; j < 4; ++j)
      acc[j] = __builtin_amdgcn_wmma_f32_16x16x32_f16(
          false, f0.a, false, f0.b[j], (short)0, acc[j], false, false);
    if (k0 + 64 < K) load_stage(f0, k0 + 64);
#pragma unroll
    for (int j = 0; j < 4; ++j)
      acc[j] = __builtin_amdgcn_wmma_f32_16x16x32_f16(
          false, f1.a, false, f1.b[j], (short)0, acc[j], false, false);
  }

  int roff = (lane >> 4) << 3;
#pragma unroll
  for (int j = 0; j < 4; ++j) {
    int n = n0 + j * 16 + lm;
    float bn = bias[n];
#pragma unroll
    for (int r = 0; r < 8; ++r) {
      size_t o = (size_t)(m0 + roff + r) * N + n;
      float v = acc[j][r] + bn;
      if (residual) v += residual[o];
      C[o] = v;
    }
  }
}

// ---------------------------------------------------------------------------
// 3x3 conv (pad 1) over (512, 200, 200) -> 192 channels, implicit GEMM.
// One wave computes a 16(position) x 64(out-channel) tile.
// Flattened k-loop: ks in [0,144), tap = ks/16, ic0 = (ks%16)*32; the B k-index
// is exactly ks*32 because weights are packed [oc][tap][ic].
// ---------------------------------------------------------------------------
__global__ __launch_bounds__(128, 1)
void conv_gemm_kernel(const _Float16* __restrict__ Q,   // (NQ, 512)
                      const _Float16* __restrict__ W,   // (192, 4608)
                      const float* __restrict__ off_b,
                      const float* __restrict__ attn_b,
                      float* __restrict__ C) {          // (NQ, 192)
  int wave = blockIdx.x * (blockDim.x >> 5) + (threadIdx.x >> 5);
  int lane = threadIdx.x & 31;
  int mt = wave / 3, ng = wave % 3;
  int m0 = mt * 16;
  if (m0 >= NQ) return;
  int n0 = ng * 64;
  int lm    = lane & 15;
  int aoff  = (lane >> 4) << 4;
  int boffb = (lane >> 4) << 4;

  int pos = m0 + lm;
  int y = pos / BEVW, x = pos % BEVW;

  const _Float16* bp0 = W + (size_t)(n0 + lm) * KCONV + boffb;
#pragma unroll
  for (int j = 0; j < 4; ++j)
    __builtin_prefetch(bp0 + (size_t)j * 16 * KCONV, 0, 0);

  // Zero-padded shifted A-row fragment + B fragments for k-step ks.
  auto load_stage = [&](Frags& f, int ks) {
    int tap = ks >> 4;
    int ic0 = (ks & 15) << 5;
    int py = y + tap / 3 - 1;
    int px = x + tap % 3 - 1;
    v16h z = {};
    if (py < 0 || py >= BEVH || px < 0 || px >= BEVW) {
      f.a = z;
    } else {
      const _Float16* rp = Q + (size_t)(py * BEVW + px) * 512 + aoff;
      f.a = load_frag16(rp, ic0, ic0 + 8);
    }
    int k = ks << 5;
#pragma unroll
    for (int j = 0; j < 4; ++j)
      f.b[j] = load_frag16(bp0 + (size_t)j * 16 * KCONV, k, k + 8);
  };

  v8f acc[4] = {};
  Frags f0, f1;
  load_stage(f0, 0);

  for (int ks = 0; ks < 144; ks += 2) {
    load_stage(f1, ks + 1);             // always in range (144 even)
#pragma unroll
    for (int j = 0; j < 4; ++j)
      acc[j] = __builtin_amdgcn_wmma_f32_16x16x32_f16(
          false, f0.a, false, f0.b[j], (short)0, acc[j], false, false);
    if (ks + 2 < 144) load_stage(f0, ks + 2);
#pragma unroll
    for (int j = 0; j < 4; ++j)
      acc[j] = __builtin_amdgcn_wmma_f32_16x16x32_f16(
          false, f1.a, false, f1.b[j], (short)0, acc[j], false, false);
  }

  int roff = (lane >> 4) << 3;
#pragma unroll
  for (int j = 0; j < 4; ++j) {
    int n = n0 + j * 16 + lm;
    float bn = (n < 128) ? off_b[n] : attn_b[n - 128];
#pragma unroll
    for (int r = 0; r < 8; ++r)
      C[(size_t)(m0 + roff + r) * CCONV + n] = acc[j][r] + bn;
  }
}

// ---------------------------------------------------------------------------
// Softmax + bilinear deformable sampling + queue average.
// One wave per (q, h); lane = d (D == 32 == wave32).  Output X in f16.
// ---------------------------------------------------------------------------
__device__ __forceinline__ float corner_sample(const float* __restrict__ valp,
                                               int b, int h, int xi, int yi,
                                               float w, int d) {
  if (xi < 0 || xi >= BEVW || yi < 0 || yi >= BEVH) return 0.0f;
  int idx = yi * BEVW + xi;
  return valp[((size_t)(b * NQ + idx)) * EDIM + h * DDIM + d] * w;
}

__global__ void sample_kernel(const float* __restrict__ valp,     // (2*NQ, 256)
                              const float* __restrict__ conv,     // (NQ, 192)
                              const float* __restrict__ pre_ref,  // (NQ, 2)
                              const float* __restrict__ cur_ref,  // (NQ, 2)
                              _Float16* __restrict__ Xh) {        // (NQ, 256)
  int wave = blockIdx.x * (blockDim.x >> 5) + (threadIdx.x >> 5);
  int lane = threadIdx.x & 31;   // d
  int q = wave >> 3;
  int h = wave & 7;
  if (q >= NQ) return;
  const float* cv = conv + (size_t)q * CCONV;
  float result = 0.0f;
#pragma unroll
  for (int b = 0; b < 2; ++b) {
    const float* ref = (b == 0) ? pre_ref : cur_ref;
    float rx = ref[q * 2 + 0] * (float)BEVW - 0.5f;
    float ry = ref[q * 2 + 1] * (float)BEVH - 0.5f;
    // softmax over the 4 points of this (h, queue)
    float a0 = cv[128 + h * 8 + b * 4 + 0];
    float a1 = cv[128 + h * 8 + b * 4 + 1];
    float a2 = cv[128 + h * 8 + b * 4 + 2];
    float a3 = cv[128 + h * 8 + b * 4 + 3];
    float m  = fmaxf(fmaxf(a0, a1), fmaxf(a2, a3));
    float e0 = __expf(a0 - m), e1 = __expf(a1 - m);
    float e2 = __expf(a2 - m), e3 = __expf(a3 - m);
    float inv = 1.0f / (e0 + e1 + e2 + e3);
    float aw[4] = {e0 * inv, e1 * inv, e2 * inv, e3 * inv};
#pragma unroll
    for (int p = 0; p < 4; ++p) {
      float ox = cv[h * 16 + b * 8 + p * 2 + 0];
      float oy = cv[h * 16 + b * 8 + p * 2 + 1];
      float xx = rx + ox;          // (ref + off/W)*W - 0.5
      float yy = ry + oy;
      float x0f = floorf(xx), y0f = floorf(yy);
      float fx = xx - x0f, fy = yy - y0f;
      int ix = (int)x0f, iy = (int)y0f;
      float s = corner_sample(valp, b, h, ix,     iy,     (1.f-fx)*(1.f-fy), lane)
              + corner_sample(valp, b, h, ix + 1, iy,     fx*(1.f-fy),       lane)
              + corner_sample(valp, b, h, ix,     iy + 1, (1.f-fx)*fy,       lane)
              + corner_sample(valp, b, h, ix + 1, iy + 1, fx*fy,             lane);
      result += aw[p] * s;
    }
  }
  Xh[(size_t)q * EDIM + h * DDIM + lane] = (_Float16)(result * 0.5f);
}

// ---------------------------------------------------------------------------
// Host launch
// ---------------------------------------------------------------------------
extern "C" void kernel_launch(void* const* d_in, const int* in_sizes, int n_in,
                              void* d_out, int out_size, void* d_ws, size_t ws_size,
                              hipStream_t stream) {
  const float* query      = (const float*)d_in[0];
  const float* query_pos  = (const float*)d_in[3];
  const float* ref_pts    = (const float*)d_in[4];
  const float* pre_bev    = (const float*)d_in[7];
  const float* pre_ref    = (const float*)d_in[8];
  const unsigned char* sos = (const unsigned char*)d_in[9];
  const float* Wv     = (const float*)d_in[10];
  const float* bv     = (const float*)d_in[11];
  const float* Wo     = (const float*)d_in[12];
  const float* bo     = (const float*)d_in[13];
  const float* off_w  = (const float*)d_in[14];
  const float* off_b  = (const float*)d_in[15];
  const float* attn_w = (const float*)d_in[16];
  const float* attn_b = (const float*)d_in[17];
  float* out = (float*)d_out;

  char* ws = (char*)d_ws;
  size_t o = 0;
  auto alloc = [&](size_t bytes) -> void* {
    o = (o + 255) & ~(size_t)255;
    void* p = ws + o;
    o += bytes;
    return p;
  };
  _Float16* val_h = (_Float16*)alloc((size_t)2 * NQ * EDIM * 2);
  _Float16* q_h   = (_Float16*)alloc((size_t)NQ * 512 * 2);
  _Float16* Wvh   = (_Float16*)alloc((size_t)EDIM * EDIM * 2);
  _Float16* Woh   = (_Float16*)alloc((size_t)EDIM * EDIM * 2);
  _Float16* Wch   = (_Float16*)alloc((size_t)CCONV * KCONV * 2);
  float*    valp  = (float*)alloc((size_t)2 * NQ * EDIM * 4);
  float*    convo = (float*)alloc((size_t)NQ * CCONV * 4);
  _Float16* Xh    = (_Float16*)alloc((size_t)NQ * EDIM * 2);

  // 1) prep
  prep_data_kernel<<<(NQ * EDIM + 255) / 256, 256, 0, stream>>>(
      query, query_pos, pre_bev, sos, val_h, q_h);
  prep_gemm_w_kernel<<<(EDIM * EDIM + 255) / 256, 256, 0, stream>>>(Wv, Wo, Wvh, Woh);
  prep_conv_w_kernel<<<(CCONV * KCONV + 255) / 256, 256, 0, stream>>>(off_w, attn_w, Wch);

  // 2) value projection: (80000 x 256) @ (256 x 256)^T + bv
  {
    int M = 2 * NQ, N = EDIM;
    int waves = (M / 16) * (N / 64);             // 20000
    gemm_kernel<<<(waves + 3) / 4, 128, 0, stream>>>(
        val_h, Wvh, bv, nullptr, valp, M, EDIM, N);
  }

  // 3) fused offset+attention conv (implicit GEMM)
  {
    int waves = (NQ / 16) * (CCONV / 64);        // 7500
    conv_gemm_kernel<<<(waves + 3) / 4, 128, 0, stream>>>(
        q_h, Wch, off_b, attn_b, convo);
  }

  // 4) softmax + bilinear sampling + queue average -> X (f16)
  {
    int waves = NQ * NH;                         // 320000
    sample_kernel<<<waves / 8, 256, 0, stream>>>(valp, convo, pre_ref, ref_pts, Xh);
  }

  // 5) output projection + bias + residual (identity = query)
  {
    int waves = (NQ / 16) * (EDIM / 64);         // 10000
    gemm_kernel<<<(waves + 3) / 4, 128, 0, stream>>>(
        Xh, Woh, bo, query, out, NQ, EDIM, EDIM);
  }
}
